// LSTM_SRU_45028437131868
// MI455X (gfx1250) — compile-verified, matrix operand order checked by reading, not run on hardware
//
#include <hip/hip_runtime.h>

typedef __attribute__((ext_vector_type(16))) __bf16 v16bf;
typedef __attribute__((ext_vector_type(8)))  __bf16 v8bf;
typedef __attribute__((ext_vector_type(8)))  float  v8f;
typedef __attribute__((ext_vector_type(4)))  unsigned int v4u;
typedef __attribute__((ext_vector_type(8)))  int v8i;
typedef __attribute__((ext_vector_type(4)))  int v4i;

#define SEQT   512
#define BATCH  64
#define HDIM   1024
#define KCOMB  2048

#if defined(__HIP_DEVICE_COMPILE__) && __has_builtin(__builtin_amdgcn_tensor_load_to_lds)
#define USE_TDM 1
#else
#define USE_TDM 0
#endif

// ---------------- fast transcendentals (v_exp_f32 / v_rcp_f32) -------------
__device__ __forceinline__ float fast_exp2(float x) {
#if defined(__HIP_DEVICE_COMPILE__) && __has_builtin(__builtin_amdgcn_exp2f)
    return __builtin_amdgcn_exp2f(x);
#else
    return exp2f(x);
#endif
}
__device__ __forceinline__ float fast_rcp(float x) {
#if defined(__HIP_DEVICE_COMPILE__) && __has_builtin(__builtin_amdgcn_rcpf)
    return __builtin_amdgcn_rcpf(x);
#else
    return 1.0f / x;
#endif
}
__device__ __forceinline__ float fast_sigmoid(float x) {
    return fast_rcp(1.0f + fast_exp2(-1.4426950408889634f * x));
}
__device__ __forceinline__ float fast_tanh(float x) {
    float e = fast_exp2(2.8853900817779268f * x);   // exp(2x)
    return 1.0f - 2.0f * fast_rcp(e + 1.0f);
}

// ---- bf16 fragment loader (CDNA5 16-bit A/B layout), generic ptr ----------
// lanes 0-15 : row = lane,    K = {k0..k0+7, k0+16..k0+23}
// lanes 16-31: row = lane-16, K = {k0+8..k0+15, k0+24..k0+31}
// Works for both LDS (ds_load_b128) and global (global_load_b128) sources.
__device__ __forceinline__ v16bf frag_bf16(const __bf16* base, int ld, int row0, int k0) {
    int lane  = threadIdx.x & 31;
    int r     = lane & 15;
    int khalf = (lane >> 4) << 3;           // 0 or 8
    const __bf16* p = base + (size_t)(row0 + r) * ld + k0 + khalf;
    v8bf lo = *(const v8bf*)p;
    v8bf hi = *(const v8bf*)(p + 16);
    v16bf f;
#pragma unroll
    for (int j = 0; j < 8; ++j) { f[j] = lo[j]; f[8 + j] = hi[j]; }
    return f;
}

__device__ __forceinline__ v8f wmma_bf16(v16bf a, v16bf b, v8f c) {
    return __builtin_amdgcn_wmma_f32_16x16x32_bf16(
        false, a, false, b, (short)0, c, false, false);
}

#if USE_TDM
// ---- Tensor Data Mover: 2-D tile global->LDS (D# per ISA 08_async_tensor §8)
// clang-23 / therock toolchain: 6-arg builtin (g0 v4u, g1 v8i, g2 v4i, g3 v4i, g4 v8i, cpol)
__device__ __forceinline__ void tdm_load_2d(const __bf16* gsrc, __bf16* ldsdst,
                                            unsigned rows, unsigned cols,
                                            unsigned stride_elems) {
    unsigned long long ga = (unsigned long long)(uintptr_t)gsrc;
    unsigned lds = (unsigned)(uintptr_t)ldsdst;   // low 32 bits = LDS offset
    v4u g0;
    g0[0] = 1u;                                   // count=1, user descriptor
    g0[1] = lds;                                  // lds_addr
    g0[2] = (unsigned)ga;                         // global_addr[31:0]
    g0[3] = (unsigned)((ga >> 32) & 0x1FFFFFFull) // global_addr[56:32]
          | (2u << 30);                           // type = 2 ("image")
    v8i g1;
    g1[0] = (int)(1u << 16);                      // data_size=1 (2 bytes), no mask
    g1[1] = (int)((cols & 0xFFFFu) << 16);        // tensor_dim0[15:0]
    g1[2] = (int)(((cols >> 16) & 0xFFFFu) | ((rows & 0xFFFFu) << 16)); // dim0 hi | dim1 lo
    g1[3] = (int)(((rows >> 16) & 0xFFFFu) | ((cols & 0xFFFFu) << 16)); // dim1 hi | tile_dim0
    g1[4] = (int)(rows & 0xFFFFu);                // tile_dim1 (tile_dim2=0)
    g1[5] = (int)stride_elems;                    // tensor_dim0_stride[31:0]
    g1[6] = 0;                                    // dim0_stride hi | dim1_stride lo
    g1[7] = 0;
    v4i z4 = {0, 0, 0, 0};
    v8i z8 = {0, 0, 0, 0, 0, 0, 0, 0};
    __builtin_amdgcn_tensor_load_to_lds(g0, g1, z4, z4, z8, 0);
}
#endif

// ---- fused LSTM/SRU cell: gates + tx GEMMs + elementwise ------------------
// grid = 32 blocks (32 output cols each), block = 256 threads = 8 waves.
// wave w: mtile = w&3 (16 batch rows), nt = w>>2 (16-col group)
__global__ __launch_bounds__(256) void lstm_cell_kernel(
    const __bf16* __restrict__ xsrc,    // (64,1024) layer input, bf16
    const __bf16* __restrict__ hin,     // (64,1024) prev hidden, bf16
    const __bf16* __restrict__ Wc,      // (4096,2048) gates weight, bf16
    const __bf16* __restrict__ Wt,      // (1024,1024) tx weight, bf16
    const float*  __restrict__ b_all,   // (4096,)
    const float*  __restrict__ b_t,     // (1024,)
    float*        __restrict__ c_state, // (64,1024) in/out fp32
    __bf16*       __restrict__ h_out_bf,// (64,1024) next hidden, bf16
    float*        __restrict__ h_out_f32,// (64,1024) next hidden, fp32
    float*        __restrict__ seq_out) // (64,1024) or nullptr
{
    int lane    = threadIdx.x & 31;
    int w       = threadIdx.x >> 5;
    int mtile   = w & 3;
    int nt      = w >> 2;
    int row0    = mtile * 16;
    int colbase = blockIdx.x * 32 + nt * 16;

#if USE_TDM
    // Stage the whole A panel (x_t | h) into LDS once via the TDM.
    extern __shared__ __bf16 smem[];            // 2 * 64*1024 bf16 = 256 KB
    if (threadIdx.x == 0) {                     // one wave issues both DMAs
        tdm_load_2d(xsrc, smem, BATCH, HDIM, HDIM);
        tdm_load_2d(hin,  smem + BATCH * HDIM, BATCH, HDIM, HDIM);
        __builtin_amdgcn_s_wait_tensorcnt(0);
    }
    __syncthreads();
    const __bf16* xA = smem;
    const __bf16* hA = smem + BATCH * HDIM;
#else
    const __bf16* xA = xsrc;
    const __bf16* hA = hin;
#endif

    v8f acc_i = {}; v8f acc_f = {}; v8f acc_o = {}; v8f acc_g = {}; v8f acc_t = {};

    // ---- K half 1: combined[:, 0:1024] == x_t  (gates + tx, shared A frag)
    for (int kc = 0; kc < HDIM; kc += 32) {
        v16bf a = frag_bf16(xA, HDIM, row0, kc);
        v16bf bi = frag_bf16(Wc, KCOMB, 0 * HDIM + colbase, kc);
        acc_i = wmma_bf16(a, bi, acc_i);
        v16bf bf = frag_bf16(Wc, KCOMB, 1 * HDIM + colbase, kc);
        acc_f = wmma_bf16(a, bf, acc_f);
        v16bf bo = frag_bf16(Wc, KCOMB, 2 * HDIM + colbase, kc);
        acc_o = wmma_bf16(a, bo, acc_o);
        v16bf bg = frag_bf16(Wc, KCOMB, 3 * HDIM + colbase, kc);
        acc_g = wmma_bf16(a, bg, acc_g);
        v16bf bt = frag_bf16(Wt, HDIM, colbase, kc);
        acc_t = wmma_bf16(a, bt, acc_t);
    }

    // ---- K half 2: combined[:, 1024:2048] == h  (gates only)
    for (int kc = 0; kc < HDIM; kc += 32) {
        v16bf a = frag_bf16(hA, HDIM, row0, kc);
        v16bf bi = frag_bf16(Wc, KCOMB, 0 * HDIM + colbase, HDIM + kc);
        acc_i = wmma_bf16(a, bi, acc_i);
        v16bf bf = frag_bf16(Wc, KCOMB, 1 * HDIM + colbase, HDIM + kc);
        acc_f = wmma_bf16(a, bf, acc_f);
        v16bf bo = frag_bf16(Wc, KCOMB, 2 * HDIM + colbase, HDIM + kc);
        acc_o = wmma_bf16(a, bo, acc_o);
        v16bf bg = frag_bf16(Wc, KCOMB, 3 * HDIM + colbase, HDIM + kc);
        acc_g = wmma_bf16(a, bg, acc_g);
    }

    int col = colbase + (lane & 15);
    float bbi = b_all[col];
    float bbf = b_all[HDIM + col];
    float bbo = b_all[2 * HDIM + col];
    float bbg = b_all[3 * HDIM + col];
    float bbt = b_t[col];

#pragma unroll
    for (int r = 0; r < 8; ++r) {
        // C/D layout: VGPR r holds M=r (lanes 0-15) and M=r+8 (lanes 16-31)
        int m = row0 + r + ((lane >> 4) << 3);
        size_t idx = (size_t)m * HDIM + col;
        float si = fast_sigmoid(acc_i[r] + bbi);
        float sf = fast_sigmoid(acc_f[r] + bbf);
        float so = fast_sigmoid(acc_o[r] + bbo);
        float gt = fast_tanh((acc_t[r] + bbt) * (acc_g[r] + bbg));
        float omf  = 1.0f - sf;
        float fmix = si * (1.0f - omf * omf) + (1.0f - si) * sf * sf;
        float cn = fmix * c_state[idx] + (1.0f - fmix) * gt;
        c_state[idx] = cn;
        float hn = so * fast_tanh(cn);
        h_out_bf[idx]  = (__bf16)hn;
        h_out_f32[idx] = hn;
        if (seq_out) seq_out[idx] = hn;
    }
}

__global__ void cvt_f32_bf16_kernel(const float* __restrict__ in,
                                    __bf16* __restrict__ out, int n) {
    int i = blockIdx.x * blockDim.x + threadIdx.x;
    if (i < n) out[i] = (__bf16)in[i];
}

extern "C" void kernel_launch(void* const* d_in, const int* in_sizes, int n_in,
                              void* d_out, int out_size, void* d_ws, size_t ws_size,
                              hipStream_t stream) {
    (void)in_sizes; (void)n_in; (void)out_size; (void)ws_size;

    const float* x      = (const float*)d_in[0];
    const float* W_all0 = (const float*)d_in[1];
    const float* b_all0 = (const float*)d_in[2];
    const float* W_t0   = (const float*)d_in[3];
    const float* b_t0   = (const float*)d_in[4];
    const float* W_all1 = (const float*)d_in[5];
    const float* b_all1 = (const float*)d_in[6];
    const float* W_t1   = (const float*)d_in[7];
    const float* b_t1   = (const float*)d_in[8];
    float* outp = (float*)d_out;

    // workspace carve-out (~105 MB)
    char* p = (char*)d_ws;
    auto take = [&](size_t bytes) -> char* {
        char* r = p; p += (bytes + 255) & ~(size_t)255; return r;
    };
    __bf16* Wb0  = (__bf16*)take((size_t)4096 * 2048 * 2);
    __bf16* Wb1  = (__bf16*)take((size_t)4096 * 2048 * 2);
    __bf16* Wtb0 = (__bf16*)take((size_t)1024 * 1024 * 2);
    __bf16* Wtb1 = (__bf16*)take((size_t)1024 * 1024 * 2);
    __bf16* xb   = (__bf16*)take((size_t)SEQT * BATCH * HDIM * 2);
    __bf16* hb0[2] = { (__bf16*)take((size_t)BATCH * HDIM * 2),
                       (__bf16*)take((size_t)BATCH * HDIM * 2) };
    __bf16* hb1[2] = { (__bf16*)take((size_t)BATCH * HDIM * 2),
                       (__bf16*)take((size_t)BATCH * HDIM * 2) };
    float* h0f = (float*)take((size_t)BATCH * HDIM * 4);
    float* h1f = (float*)take((size_t)BATCH * HDIM * 4);
    float* c0  = (float*)take((size_t)BATCH * HDIM * 4);
    float* c1  = (float*)take((size_t)BATCH * HDIM * 4);

    // fp32 -> bf16 conversions (weights stay L2-resident: 36 MB << 192 MB)
    const int nWall = 4096 * 2048, nWt = 1024 * 1024, nX = SEQT * BATCH * HDIM;
    cvt_f32_bf16_kernel<<<(nWall + 255) / 256, 256, 0, stream>>>(W_all0, Wb0, nWall);
    cvt_f32_bf16_kernel<<<(nWall + 255) / 256, 256, 0, stream>>>(W_all1, Wb1, nWall);
    cvt_f32_bf16_kernel<<<(nWt + 255) / 256, 256, 0, stream>>>(W_t0, Wtb0, nWt);
    cvt_f32_bf16_kernel<<<(nWt + 255) / 256, 256, 0, stream>>>(W_t1, Wtb1, nWt);
    cvt_f32_bf16_kernel<<<(nX + 255) / 256, 256, 0, stream>>>(x, xb, nX);

    (void)hipMemsetAsync(hb0[0], 0, (size_t)BATCH * HDIM * 2, stream);
    (void)hipMemsetAsync(hb1[0], 0, (size_t)BATCH * HDIM * 2, stream);
    (void)hipMemsetAsync(c0, 0, (size_t)BATCH * HDIM * 4, stream);
    (void)hipMemsetAsync(c1, 0, (size_t)BATCH * HDIM * 4, stream);

    const size_t smemBytes = (size_t)2 * BATCH * HDIM * 2;  // 256 KB (<=320 KB WGP LDS)
    for (int t = 0; t < SEQT; ++t) {
        const __bf16* h0p = hb0[t & 1];
        __bf16*       h0n = hb0[(t + 1) & 1];
        const __bf16* h1p = hb1[t & 1];
        __bf16*       h1n = hb1[(t + 1) & 1];
        lstm_cell_kernel<<<dim3(HDIM / 32), dim3(256), smemBytes, stream>>>(
            xb + (size_t)t * BATCH * HDIM, h0p, Wb0, Wtb0, b_all0, b_t0,
            c0, h0n, h0f, nullptr);
        lstm_cell_kernel<<<dim3(HDIM / 32), dim3(256), smemBytes, stream>>>(
            h0n, h1p, Wb1, Wtb1, b_all1, b_t1,
            c1, h1n, h1f, outp + (size_t)t * BATCH * HDIM);
    }

    // outputs | h_final(2,64,1024) | c_final(2,64,1024)
    const size_t stateBytes = (size_t)BATCH * HDIM * 4;
    float* tail = outp + (size_t)SEQT * BATCH * HDIM;
    (void)hipMemcpyAsync(tail,                            h0f, stateBytes, hipMemcpyDeviceToDevice, stream);
    (void)hipMemcpyAsync(tail + (size_t)BATCH * HDIM,     h1f, stateBytes, hipMemcpyDeviceToDevice, stream);
    (void)hipMemcpyAsync(tail + (size_t)2 * BATCH * HDIM, c0,  stateBytes, hipMemcpyDeviceToDevice, stream);
    (void)hipMemcpyAsync(tail + (size_t)3 * BATCH * HDIM, c1,  stateBytes, hipMemcpyDeviceToDevice, stream);
}
